// FunkNN_58222576664924
// MI455X (gfx1250) — matrix operand order repaired and checked by reading, not compiled, not run on hardware
//
#include <hip/hip_runtime.h>
#include <hip/hip_bf16.h>

typedef unsigned int   u32;
typedef unsigned short u16;
typedef int v4i __attribute__((__vector_size__(16)));
typedef __attribute__((address_space(1))) v4i* gv4i_p;   // global
typedef __attribute__((address_space(3))) v4i* lv4i_p;   // LDS
typedef __attribute__((ext_vector_type(16))) __bf16 v16bf;
typedef __attribute__((ext_vector_type(8)))  float  v8f;

// ---------------------------------------------------------------- constants
#define ROWS   2048            // B * B_PIXELS
#define CANG   180             // projection angles
#define NDET   183             // detector bins
#define PADN   512             // FFT pad size
#define K1     14580           // C * 9 * 9
#define K1P    14592           // K1 padded to multiple of 64 (228*64)

// --------- feature probes for gfx1250-specific data movement (device pass)
#ifdef __has_builtin
#if __has_builtin(__builtin_amdgcn_global_load_async_to_lds_b128)
#define USE_ASYNC 1
#endif
#if __has_builtin(__builtin_amdgcn_s_wait_asynccnt)
#define WAIT_ASYNC(n) __builtin_amdgcn_s_wait_asynccnt(n)
#endif
#endif
#ifndef USE_ASYNC
#define USE_ASYNC 0
#endif
#ifndef WAIT_ASYNC
#define WAIT_ASYNC(n) asm volatile("s_wait_asynccnt %0" :: "i"(n) : "memory")
#endif

#if USE_ASYNC
#define NBUF 3
#else
#define NBUF 2
#endif

__device__ __forceinline__ u16 f2bf(float f) {           // f32 -> bf16 RNE
  u32 u = __float_as_uint(f);
  u32 r = u + 0x7FFFu + ((u >> 16) & 1u);
  return (u16)(r >> 16);
}
__device__ __forceinline__ float bf2f(u16 h) {
  return __uint_as_float(((u32)h) << 16);
}

// ------------------------------------------------- 1) real ifft of ramp filter
__global__ void k_make_h(const float* __restrict__ F, float* __restrict__ h) {
  int m = threadIdx.x;                       // 512 threads, 1 block
  float s = 0.f;
  const float w = 6.2831853071795864769f / 512.0f * (float)m;
  for (int k = 0; k < PADN; ++k) s += F[k] * __cosf(w * (float)k);
  h[m] = s * (1.0f / 512.0f);
}

// --------------------------------- 2) circular conv == FFT*filter*IFFT (real)
__global__ void k_filter(const float* __restrict__ sino,   // (B, NDET, CANG)
                         const float* __restrict__ h,      // (512,)
                         float* __restrict__ col) {        // (B, CANG, NDET)
  __shared__ float s_h[PADN];
  __shared__ float s_col[NDET];
  const int bid = blockIdx.x;                // b*CANG + c
  const int b = bid / CANG, c = bid - b * CANG;
  for (int k = threadIdx.x; k < PADN; k += blockDim.x) s_h[k] = h[k];
  for (int j = threadIdx.x; j < NDET; j += blockDim.x)
    s_col[j] = sino[((size_t)b * NDET + j) * CANG + c];
  __syncthreads();
  const int i = threadIdx.x;
  if (i < NDET) {
    float s = 0.f;
    for (int j = 0; j < NDET; ++j) s += s_col[j] * s_h[(i - j + PADN) & (PADN - 1)];
    col[(size_t)bid * NDET + i] = s;
  }
}

// -------------------- 3) grid build + reflect + rotate + 1D sample -> X bf16
__global__ __launch_bounds__(256)
void k_sample(const float* __restrict__ coord,   // (B,512,2)
              const float* __restrict__ col,     // (B,CANG,NDET)
              const float* __restrict__ ws1, const float* __restrict__ ws2,
              const float* __restrict__ sp, const float* __restrict__ theta,
              u16* __restrict__ X) {             // (ROWS, K1P) bf16
  const int row = blockIdx.x;                          // 0..2047
  const int f   = blockIdx.y * 256 + threadIdx.x;      // 0..14591
  const int b = row >> 9, p = row & 511;
  float val = 0.f;
  if (f < K1) {
    const int c  = f / 81;
    const int r2 = f - c * 81;
    const int iy = r2 / 9, ix = r2 - iy * 9;
    const float c0 = coord[((size_t)b * 512 + p) * 2 + 0];
    const float c1 = coord[((size_t)b * 512 + p) * 2 + 1];
    // crop/2 = (w-1)/(h-1) = 8/128 ; lin[i] = (i-4)/4
    const float gx = ws1[0] * 0.0625f * ((float)(ix - 4) * 0.25f) + 2.f * c1;
    const float gy = ws2[0] * 0.0625f * ((float)(iy - 4) * 0.25f) + 2.f * c0;
    // coords = (gy, gx)/2 ; reflect((v+0.5)*128, -0.5, 128.5) ; recentre -64
    float uy = (gy * 0.5f + 0.5f) * 128.f;
    float ux = (gx * 0.5f + 0.5f) * 128.f;
    uy = (uy > 128.5f) ? (257.f - uy) : ((uy < -0.5f) ? (-1.f - uy) : uy);
    ux = (ux > 128.5f) ? (257.f - ux) : ((ux < -0.5f) ? (-1.f - ux) : ux);
    const float scale = sp[0] * (182.f / 182.f);       // n == n_base
    const float xpr = (uy - 64.f) / scale;             // component 0 (gy)
    const float ypr = (ux - 64.f) / scale;             // component 1 (gx)
    const float th = theta[c] * 0.017453292519943295f;
    float st, ct; __sincosf(th, &st, &ct);
    const float t = ypr * ct - xpr * st;
    // z maps exactly back to angle row c -> 1D interp along detector axis
    const float x  = (t + 1.f) * 91.f;                 // (t+1)*0.5*(NDET-1)
    const float x0f = floorf(x);
    const float wx  = x - x0f;
    const int   x0  = (int)x0f;
    const float* rp = col + ((size_t)b * CANG + c) * NDET;
    const float v0 = (x0 >= 0 && x0 <= NDET - 1) ? rp[x0] : 0.f;
    const float v1 = (x0 + 1 >= 0 && x0 + 1 <= NDET - 1) ? rp[x0 + 1] : 0.f;
    val = v0 * (1.f - wx) + v1 * wx;
  }
  X[(size_t)row * K1P + f] = f2bf(val);
}

// ------------------- 4) weight f32 (N,K) -> bf16 transposed (Kp,N), 0-padded
__global__ __launch_bounds__(256)
void k_convw_t(const float* __restrict__ W, u16* __restrict__ O,
               int N, int K) {
  __shared__ u16 tile[64][33];               // [n][k], padded k-stride
  const int t = threadIdx.x;
  const int k0 = blockIdx.x * 32, n0 = blockIdx.y * 64;
  {
    const int n = t >> 2;                    // 0..63
    const int kc = (t & 3) * 8;              // 0,8,16,24
    #pragma unroll
    for (int j = 0; j < 8; ++j) {
      const int k = k0 + kc + j;
      tile[n][kc + j] = (k < K) ? f2bf(W[(size_t)(n0 + n) * K + k]) : (u16)0;
    }
  }
  __syncthreads();
  {
    const int r = t >> 3;                    // k row 0..31
    const int c = (t & 7) * 8;               // n col 0..56
    union { u16 h[8]; uint4 q; } v;
    #pragma unroll
    for (int j = 0; j < 8; ++j) v.h[j] = tile[c + j][r];
    *(uint4*)&O[(size_t)(k0 + r) * N + n0 + c] = v.q;
  }
}

// ---------------- 5) bf16 WMMA GEMM: Out = relu(A(M,Kp) * Wt(Kp,N) + bias)
// Block tile 128x64, BK=64 (two WMMA sub-steps per tile), 8 waves each owning
// 32x32 -> 8 v_wmma_f32_16x16x32_bf16 per tile per wave, one barrier per tile.
// Async path: triple-buffered LDS with two tiles of DMA in flight
// (s_wait_asynccnt <= 6 retires the oldest tile while the newest still flies).
__global__ __launch_bounds__(256)
void k_gemm_bf16(const u16* __restrict__ A, int Kp,
                 const u16* __restrict__ Wt, const float* __restrict__ bias,
                 u16* __restrict__ Out, int N, int relu) {
  __shared__ __align__(16) u16 Alds[NBUF][128 * 64];  // row-major, 8-half XOR swizzle
  __shared__ __align__(16) u16 Blds[NBUF][64 * 64];   // k-major, 16-half XOR swizzle
  const int t = threadIdx.x;
  const int lane = t & 31, wave = t >> 5;
  const int wm = wave & 3, wn = wave >> 2;
  const int m0 = blockIdx.x * 128, n0 = blockIdx.y * 64;
  const int lh = lane & 15, lq = lane >> 4;

  // cooperative copy mapping: A = 4 b128 per thread, B = 2 b128 per thread
  size_t gA[4]; int dA[4];
  #pragma unroll
  for (int s = 0; s < 4; ++s) {
    const int idx = t + s * 256;
    const int r = idx >> 3, c = (idx & 7) * 8;
    gA[s] = (size_t)(m0 + r) * Kp + c;
    dA[s] = r * 64 + (c ^ ((r & 7) * 8));
  }
  size_t gB[2]; int dB[2];
  #pragma unroll
  for (int s = 0; s < 2; ++s) {
    const int idx = t + s * 256;
    const int r = idx >> 3, c = (idx & 7) * 8;
    gB[s] = (size_t)r * N + n0 + c;                   // + kk*N per tile
    dB[s] = r * 64 + (c ^ ((r & 3) * 16));
  }

  // fragment addressing
  const int r0  = wm * 32 + lh;                       // A rows r0, r0+16
  const int rsw = (r0 & 7) * 8;                       // (r0+16)&7 == r0&7
  const int bsw = (lane & 3) * 16;
  const int nbx0 = (wn * 32) ^ bsw;
  const int nbx1 = (wn * 32 + 16) ^ bsw;

  union U { uint4 q[2]; v16bf v; };
  v8f acc[2][2] = {};
  const int nK = Kp >> 6;

  auto compute = [&](int buf) {
    #pragma unroll
    for (int ss = 0; ss < 2; ++ss) {
      const int kb = ss * 32;
      U a0, a1, b0, b1;
      a0.q[0] = *(const uint4*)&Alds[buf][r0 * 64 + ((kb + lq * 8) ^ rsw)];
      a0.q[1] = *(const uint4*)&Alds[buf][r0 * 64 + ((kb + 16 + lq * 8) ^ rsw)];
      a1.q[0] = *(const uint4*)&Alds[buf][(r0 + 16) * 64 + ((kb + lq * 8) ^ rsw)];
      a1.q[1] = *(const uint4*)&Alds[buf][(r0 + 16) * 64 + ((kb + 16 + lq * 8) ^ rsw)];
      #pragma unroll
      for (int s = 0; s < 16; ++s) {
        const int k = kb + ((s & 8) << 1) + lq * 8 + (s & 7);
        const int ksw = (k & 3) * 16;
        b0.v[s] = __builtin_bit_cast(__bf16, Blds[buf][k * 64 + ((wn * 32 + lh) ^ ksw)]);
        b1.v[s] = __builtin_bit_cast(__bf16, Blds[buf][k * 64 + ((wn * 32 + 16 + lh) ^ ksw)]);
      }
      acc[0][0] = __builtin_amdgcn_wmma_f32_16x16x32_bf16(false, a0.v, false, b0.v, (short)0, acc[0][0], false, false);
      acc[0][1] = __builtin_amdgcn_wmma_f32_16x16x32_bf16(false, a0.v, false, b1.v, (short)0, acc[0][1], false, false);
      acc[1][0] = __builtin_amdgcn_wmma_f32_16x16x32_bf16(false, a1.v, false, b0.v, (short)0, acc[1][0], false, false);
      acc[1][1] = __builtin_amdgcn_wmma_f32_16x16x32_bf16(false, a1.v, false, b1.v, (short)0, acc[1][1], false, false);
    }
  };

#if USE_ASYNC
  // ------- ASYNCcnt-tracked LDS DMA pipeline, 2 tiles in flight -------
  auto issue = [&](int buf, int kk) {
    #pragma unroll
    for (int s = 0; s < 4; ++s)
      __builtin_amdgcn_global_load_async_to_lds_b128(
          (gv4i_p)(const void*)(A + gA[s] + kk),
          (lv4i_p)(void*)&Alds[buf][dA[s]], 0, 0);
    #pragma unroll
    for (int s = 0; s < 2; ++s)
      __builtin_amdgcn_global_load_async_to_lds_b128(
          (gv4i_p)(const void*)(Wt + (size_t)kk * N + gB[s]),
          (lv4i_p)(void*)&Blds[buf][dB[s]], 0, 0);
  };
  issue(0, 0);
  if (nK > 1) { issue(1, 64); WAIT_ASYNC(6); }
  else        { WAIT_ASYNC(0); }
  __syncthreads();
  for (int ki = 0; ki < nK; ++ki) {
    const int buf = ki % 3;
    const bool deep = (ki + 2 < nK);
    if (deep) issue((ki + 2) % 3, (ki + 2) << 6);
    compute(buf);
    if (deep) WAIT_ASYNC(6);                 // tile ki+1 retired, ki+2 in flight
    else      WAIT_ASYNC(0);                 // drain tail
    __syncthreads();
  }
#else
  // ------- register-staged double-buffer fallback -------
  uint4 ra[4], rb[2];
  auto gload = [&](int kk) {
    #pragma unroll
    for (int s = 0; s < 4; ++s) ra[s] = *(const uint4*)(A + gA[s] + kk);
    #pragma unroll
    for (int s = 0; s < 2; ++s) rb[s] = *(const uint4*)(Wt + (size_t)kk * N + gB[s]);
  };
  auto lstore = [&](int buf) {
    #pragma unroll
    for (int s = 0; s < 4; ++s) *(uint4*)&Alds[buf][dA[s]] = ra[s];
    #pragma unroll
    for (int s = 0; s < 2; ++s) *(uint4*)&Blds[buf][dB[s]] = rb[s];
  };
  gload(0);
  lstore(0);
  __syncthreads();
  for (int ki = 0; ki < nK; ++ki) {
    const int buf = ki & 1;
    const bool more = (ki + 1 < nK);
    if (more) gload((ki + 1) << 6);
    if (ki + 2 < nK) {
      __builtin_prefetch(A + gA[0] + ((size_t)(ki + 2) << 6), 0, 1);
      __builtin_prefetch(Wt + (size_t)((ki + 2) << 6) * N + gB[0], 0, 1);
    }
    compute(buf);
    if (more) lstore(buf ^ 1);
    __syncthreads();
  }
#endif

  // epilogue (D layout: lane = N, VGPR r = M row; lanes>=16 -> M+8)
  #pragma unroll
  for (int i = 0; i < 2; ++i) {
    #pragma unroll
    for (int j = 0; j < 2; ++j) {
      const int n = n0 + wn * 32 + j * 16 + lh;
      const float bv = bias[n];
      #pragma unroll
      for (int r = 0; r < 8; ++r) {
        const int m = m0 + wm * 32 + i * 16 + lq * 8 + r;
        float v = acc[i][j][r] + bv;
        if (relu) v = fmaxf(v, 0.f);
        Out[(size_t)m * N + n] = f2bf(v);
      }
    }
  }
}

// ----------------------------------------------------- 6) final 64 -> 1 layer
__global__ void k_final(const u16* __restrict__ act, const float* __restrict__ Wl,
                        const float* __restrict__ bl, float* __restrict__ out) {
  const int r = blockIdx.x * blockDim.x + threadIdx.x;
  if (r >= ROWS) return;
  float s = bl[0];
  #pragma unroll
  for (int k = 0; k < 64; ++k) s += bf2f(act[(size_t)r * 64 + k]) * Wl[k];
  out[r] = s;
}

// ============================================================== host launcher
extern "C" void kernel_launch(void* const* d_in, const int* in_sizes, int n_in,
                              void* d_out, int out_size, void* d_ws, size_t ws_size,
                              hipStream_t stream) {
  // Input index map: dict-insertion order by default, jax-alphabetical fallback.
  int iCoord = 0, iSino = 1, iFour = 2, iWs1 = 3, iWs2 = 4, iS = 5,
      iW0 = 6, iB0 = 16, iTheta = 26;
  if (n_in >= 27 && in_sizes[1] != 4 * NDET * CANG) {
    iCoord = 0; iW0 = 1; iB0 = 11; iFour = 21; iS = 22;
    iWs1 = 23; iWs2 = 24; iSino = 25; iTheta = 26;
  }
  const float* coord = (const float*)d_in[iCoord];
  const float* sino  = (const float*)d_in[iSino];
  const float* four  = (const float*)d_in[iFour];
  const float* ws1   = (const float*)d_in[iWs1];
  const float* ws2   = (const float*)d_in[iWs2];
  const float* sprm  = (const float*)d_in[iS];
  const float* theta = (const float*)d_in[iTheta];

  static const int NL[10] = {512, 512, 256, 256, 128, 128, 64, 64, 64, 1};
  static const int KL[10] = {K1, 512, 512, 256, 256, 128, 128, 64, 64, 64};
  static const int KP[10] = {K1P, 512, 512, 256, 256, 128, 128, 64, 64, 64};

  char* ws = (char*)d_ws;
  size_t off = 0;
  auto take = [&](size_t bytes) -> void* {
    void* p = ws + off;
    off = (off + bytes + 255) & ~(size_t)255;
    return p;
  };
  float* d_h   = (float*)take(PADN * sizeof(float));
  float* d_col = (float*)take((size_t)4 * CANG * NDET * sizeof(float));
  u16*   d_X   = (u16*)take((size_t)ROWS * K1P * sizeof(u16));
  u16*   d_wt[9];
  for (int l = 0; l < 9; ++l) d_wt[l] = (u16*)take((size_t)NL[l] * KP[l] * sizeof(u16));
  u16* d_actA = (u16*)take((size_t)ROWS * 512 * sizeof(u16));
  u16* d_actB = (u16*)take((size_t)ROWS * 512 * sizeof(u16));

  // 1) filter kernel taps, 2) apply circular convolution
  k_make_h<<<1, PADN, 0, stream>>>(four, d_h);
  k_filter<<<4 * CANG, 192, 0, stream>>>(sino, d_h, d_col);
  // 3) sampled feature matrix (ROWS x K1P) bf16
  k_sample<<<dim3(ROWS, K1P / 256), 256, 0, stream>>>(coord, d_col, ws1, ws2,
                                                      sprm, theta, d_X);
  // 4) weight convert + transpose to k-major bf16
  for (int l = 0; l < 9; ++l) {
    dim3 grid(KP[l] / 32, NL[l] / 64);
    k_convw_t<<<grid, 256, 0, stream>>>((const float*)d_in[iW0 + l],
                                        d_wt[l], NL[l], KL[l]);
  }
  // 5) WMMA MLP chain
  const u16* cur = d_X;
  int curKp = K1P;
  u16* bufs[2] = {d_actA, d_actB};
  for (int l = 0; l < 9; ++l) {
    u16* outb = bufs[l & 1];
    dim3 grid(ROWS / 128, NL[l] / 64);
    k_gemm_bf16<<<grid, 256, 0, stream>>>(cur, curKp, d_wt[l],
                                          (const float*)d_in[iB0 + l],
                                          outb, NL[l], 1);
    cur = outb;
    curKp = NL[l];
  }
  // 6) final linear layer -> f32 output (ROWS,)
  k_final<<<(ROWS + 255) / 256, 256, 0, stream>>>(cur, (const float*)d_in[iW0 + 9],
                                                  (const float*)d_in[iB0 + 9],
                                                  (float*)d_out);
}
